// BipertiteGraphConvolution_89395449299805
// MI455X (gfx1250) — matrix operand order repaired and checked by reading, not compiled, not run on hardware
//
#include <hip/hip_runtime.h>

#define N_U 8192
#define N_V 16384
#define DF  32
#define KS_A 16  // K splits for the u-side (K = N_V)
#define KS_B 8   // K splits for the v-side (K = N_U)

typedef __attribute__((ext_vector_type(16))) _Float16     v16h;
typedef __attribute__((ext_vector_type(8)))  float        v8f;
typedef __attribute__((ext_vector_type(2)))  int          i32x2;
typedef __attribute__((ext_vector_type(4)))  unsigned int u32x4;

union FragH {
    v16h         h;
    unsigned int u[8];
    u32x4        q[2];
};

// ---------------------------------------------------------------------------
// Kernel 1: convert+transpose features to f16.
//   vT[d][j] = (f16) v_feature[j][d]   (32 x 16384)
//   uT[d][j] = (f16) u_feature[j][d]   (32 x 8192)
// Exactly 32*(N_V+N_U) = 786432 threads.
// ---------------------------------------------------------------------------
__global__ __launch_bounds__(256) void convert_kernel(
    const float* __restrict__ u_f, const float* __restrict__ v_f,
    _Float16* __restrict__ vT, _Float16* __restrict__ uT)
{
    const int t = blockIdx.x * 256 + threadIdx.x;
    const int NVT = DF * N_V;
    if (t < NVT) {
        const int d = t / N_V, j = t - d * N_V;
        vT[t] = (_Float16)v_f[(size_t)j * DF + d];
    } else {
        const int t2 = t - NVT;
        const int d = t2 / N_U, j = t2 - d * N_U;
        uT[t2] = (_Float16)u_f[(size_t)j * DF + d];
    }
}

// ---------------------------------------------------------------------------
// Kernel 2: aggU partial = edge[16-row tile, Kslice] @ v_feature (f16 WMMA,
// f32 accumulate) + row degrees via popcount. As a BYPRODUCT, emits the packed
// edge bit-matrix bitsW[vblk][u] (vblk = v/32), bit p of bitsW[vblk][u] =
// edge[u][32*vblk+p]. Stores are coalesced (16 consecutive u32 per half-wave).
//
// A layout (16-bit 16x32): lane(half,m): VGPR i holds K = 2i + (i>=4?8:0) + 8*half.
// B layout (16-bit 32x16): lane(half,n): K = 8*half + [0..7], then 16 + 8*half + [0..7].
// C layout: lane(half,n): VGPR r = C[r + 8*half][n].
// Bit word built with compile-time masks, one variable shift per k-step.
// ---------------------------------------------------------------------------
__global__ __launch_bounds__(256) void agg_u_kernel(
    const int* __restrict__ adj, const _Float16* __restrict__ vT,
    float* __restrict__ aggU, float* __restrict__ degU,
    unsigned int* __restrict__ bitsW /* may be nullptr */)
{
    const int lane = threadIdx.x & 31;
    const int half = lane >> 4;
    const int hs   = half << 3;                             // 8*half
    const int ln   = lane & 15;
    const int tile = blockIdx.x * 8 + (threadIdx.x >> 5);   // 0..511
    const int rb   = tile << 4;
    const int kPer = N_V / KS_A;
    const int kBeg = blockIdx.y * kPer;
    const int kEnd = kBeg + kPer;

    const int    row    = rb + ln;
    const size_t rowOff = (size_t)row * N_V;
    const _Float16* b0p = vT + (size_t)ln * N_V;         // feature cols 0..15
    const _Float16* b1p = vT + (size_t)(ln + 16) * N_V;  // feature cols 16..31

    v8f c0 = {}; v8f c1 = {};
    int degI = 0;

    for (int k0 = kBeg; k0 < kEnd; k0 += 32) {
        FragH A, B0, B1;
        unsigned int bwc = 0u;
#pragma unroll
        for (int i = 0; i < 8; ++i) {
            const int posc = 2 * i + (i >= 4 ? 8 : 0);   // compile-time constant
            const i32x2 e = *(const i32x2*)(adj + rowOff + k0 + posc + hs);
            const bool b0 = e.x > 0, b1 = e.y > 0;
            A.u[i] = (b0 ? 0x3C00u : 0u) | (b1 ? 0x3C000000u : 0u);
            bwc |= (b0 ? (1u << posc) : 0u) | (b1 ? (2u << posc) : 0u);
        }
        // full 32-bit row word (both halves end up with the same value)
        const unsigned int bw = bwc << hs;                // single variable shift
        const unsigned int rw = bw | __shfl_xor(bw, 16, 32);
        if (half == 0) {
            degI += __popc(rw);
            if (bitsW) bitsW[(size_t)(k0 >> 5) * N_U + row] = rw;
        }
        const int kB = k0 + 8 * half;
        B0.q[0] = *(const u32x4*)(b0p + kB);
        B0.q[1] = *(const u32x4*)(b0p + kB + 16);
        B1.q[0] = *(const u32x4*)(b1p + kB);
        B1.q[1] = *(const u32x4*)(b1p + kB + 16);
        if (k0 + 256 < kEnd)
            __builtin_prefetch(adj + rowOff + k0 + 256, 0, 1);
        c0 = __builtin_amdgcn_wmma_f32_16x16x32_f16(false, A.h, false, B0.h,
                                                    (short)0, c0, false, false);
        c1 = __builtin_amdgcn_wmma_f32_16x16x32_f16(false, A.h, false, B1.h,
                                                    (short)0, c1, false, false);
    }

    if (half == 0) atomicAdd(degU + row, (float)degI);

#pragma unroll
    for (int r = 0; r < 8; ++r) {
        const int orow = rb + r + 8 * half;
        atomicAdd(aggU + (size_t)orow * DF + ln,      c0[r]);
        atomicAdd(aggU + (size_t)orow * DF + 16 + ln, c1[r]);
    }
}

// ---------------------------------------------------------------------------
// Kernel 3a (fast path): aggV partial from the packed bit matrix.
// Reads 16 MB of bitsW instead of 512 MB of adj. The word addresses are
// uniform per half-wave: base index forced through readfirstlane so the compiler can
// scalarize the loads (SMEM / constant cache).
// ---------------------------------------------------------------------------
__global__ __launch_bounds__(256) void agg_v_bits_kernel(
    const unsigned int* __restrict__ bitsW, const _Float16* __restrict__ uT,
    float* __restrict__ aggV, float* __restrict__ degV)
{
    const int lane = threadIdx.x & 31;
    const int half = lane >> 4;
    const int ln   = lane & 15;
    const int tile = __builtin_amdgcn_readfirstlane(
                         blockIdx.x * 8 + (threadIdx.x >> 5));   // 0..1023, uniform
    const int cb   = tile << 4;
    const int kPer = N_U / KS_B;
    const int kBeg = blockIdx.y * kPer;
    const int kEnd = kBeg + kPer;

    const int col = cb + ln;
    const int bp  = ln + (cb & 16);                  // bit position in word
    const int* wb = (const int*)(bitsW + (size_t)(cb >> 5) * N_U);
    const _Float16* b0p = uT + (size_t)ln * N_U;
    const _Float16* b1p = uT + (size_t)(ln + 16) * N_U;

    v8f c0 = {}; v8f c1 = {};
    int degI = 0;

    for (int k0 = kBeg; k0 < kEnd; k0 += 32) {
        FragH A, B0, B1;
#pragma unroll
        for (int i = 0; i < 8; ++i) {
            const int kA = k0 + 2 * i + (i >= 4 ? 8 : 0) + 8 * half;
            const i32x2 w = *(const i32x2*)(wb + kA);    // address uniform per half-wave
            const unsigned int b0 = ((unsigned int)w.x >> bp) & 1u;
            const unsigned int b1 = ((unsigned int)w.y >> bp) & 1u;
            A.u[i] = (b0 ? 0x3C00u : 0u) | (b1 ? 0x3C000000u : 0u);
            degI += (int)(b0 + b1);
        }
        const int kB = k0 + 8 * half;
        B0.q[0] = *(const u32x4*)(b0p + kB);
        B0.q[1] = *(const u32x4*)(b0p + kB + 16);
        B1.q[0] = *(const u32x4*)(b1p + kB);
        B1.q[1] = *(const u32x4*)(b1p + kB + 16);
        c0 = __builtin_amdgcn_wmma_f32_16x16x32_f16(false, A.h, false, B0.h,
                                                    (short)0, c0, false, false);
        c1 = __builtin_amdgcn_wmma_f32_16x16x32_f16(false, A.h, false, B1.h,
                                                    (short)0, c1, false, false);
    }

    const int degFull = degI + __shfl_xor(degI, 16, 32);
    if (half == 0) atomicAdd(degV + col, (float)degFull);

#pragma unroll
    for (int r = 0; r < 8; ++r) {
        const int orow = cb + r + 8 * half;
        atomicAdd(aggV + (size_t)orow * DF + ln,      c0[r]);
        atomicAdd(aggV + (size_t)orow * DF + 16 + ln, c1[r]);
    }
}

// ---------------------------------------------------------------------------
// Kernel 3b (fallback if ws too small for bitsW): aggV directly from adj.
// ---------------------------------------------------------------------------
__global__ __launch_bounds__(256) void agg_v_kernel(
    const int* __restrict__ adj, const _Float16* __restrict__ uT,
    float* __restrict__ aggV, float* __restrict__ degV)
{
    const int lane = threadIdx.x & 31;
    const int half = lane >> 4;
    const int ln   = lane & 15;
    const int tile = blockIdx.x * 8 + (threadIdx.x >> 5);   // 0..1023
    const int cb   = tile << 4;
    const int kPer = N_U / KS_B;
    const int kBeg = blockIdx.y * kPer;
    const int kEnd = kBeg + kPer;

    const int col = cb + ln;
    const _Float16* b0p = uT + (size_t)ln * N_U;
    const _Float16* b1p = uT + (size_t)(ln + 16) * N_U;

    v8f c0 = {}; v8f c1 = {};
    int degI = 0;

    for (int k0 = kBeg; k0 < kEnd; k0 += 32) {
        FragH A, B0, B1;
#pragma unroll
        for (int i = 0; i < 8; ++i) {
            const int kA = k0 + 2 * i + (i >= 4 ? 8 : 0) + 8 * half;
            const int e0 = adj[(size_t)kA * N_V + col];
            const int e1 = adj[(size_t)(kA + 1) * N_V + col];
            const bool b0 = e0 > 0, b1 = e1 > 0;
            A.u[i] = (b0 ? 0x3C00u : 0u) | (b1 ? 0x3C000000u : 0u);
            degI += (int)b0 + (int)b1;
        }
        const int kB = k0 + 8 * half;
        B0.q[0] = *(const u32x4*)(b0p + kB);
        B0.q[1] = *(const u32x4*)(b0p + kB + 16);
        B1.q[0] = *(const u32x4*)(b1p + kB);
        B1.q[1] = *(const u32x4*)(b1p + kB + 16);
        if (k0 + 64 < kEnd)
            __builtin_prefetch(adj + (size_t)(k0 + 64) * N_V + col, 0, 1);
        c0 = __builtin_amdgcn_wmma_f32_16x16x32_f16(false, A.h, false, B0.h,
                                                    (short)0, c0, false, false);
        c1 = __builtin_amdgcn_wmma_f32_16x16x32_f16(false, A.h, false, B1.h,
                                                    (short)0, c1, false, false);
    }

    const int degFull = degI + __shfl_xor(degI, 16, 32);
    if (half == 0) atomicAdd(degV + col, (float)degFull);

#pragma unroll
    for (int r = 0; r < 8; ++r) {
        const int orow = cb + r + 8 * half;
        atomicAdd(aggV + (size_t)orow * DF + ln,      c0[r]);
        atomicAdd(aggV + (size_t)orow * DF + 16 + ln, c1[r]);
    }
}

// ---------------------------------------------------------------------------
// Kernel 4: out = relu( (1/deg) * (agg @ W) ), both sides. One thread per
// output element; 32 lanes share a row (agg loads broadcast, W coalesced).
// ---------------------------------------------------------------------------
__global__ __launch_bounds__(256) void finalize_kernel(
    const float* __restrict__ aggU, const float* __restrict__ aggV,
    const float* __restrict__ degU, const float* __restrict__ degV,
    const float* __restrict__ wU,   const float* __restrict__ wV,
    float* __restrict__ out)
{
    const int t   = blockIdx.x * 256 + threadIdx.x;
    const int o   = t & 31;
    const int row = t >> 5;                 // 0 .. N_U+N_V-1

    const float* agg; const float* W; float inv; float* dst;
    if (row < N_U) {
        agg = aggU + (size_t)row * DF;
        inv = 1.0f / degU[row];
        W   = wU;
        dst = out + (size_t)row * DF + o;
    } else {
        const int r = row - N_U;
        agg = aggV + (size_t)r * DF;
        inv = 1.0f / degV[r];
        W   = wV;
        dst = out + (size_t)N_U * DF + (size_t)r * DF + o;
    }

    float acc = 0.0f;
#pragma unroll
    for (int d = 0; d < DF; ++d)
        acc += agg[d] * W[d * DF + o];
    acc *= inv;
    *dst = fmaxf(acc, 0.0f);
}

// ---------------------------------------------------------------------------
extern "C" void kernel_launch(void* const* d_in, const int* in_sizes, int n_in,
                              void* d_out, int out_size, void* d_ws, size_t ws_size,
                              hipStream_t stream)
{
    const int*   adj = (const int*)  d_in[0];
    const float* u_f = (const float*)d_in[1];
    const float* v_f = (const float*)d_in[2];
    const float* wU  = (const float*)d_in[3];
    const float* wV  = (const float*)d_in[4];
    float*       out = (float*)d_out;

    // Workspace layout (16B-aligned slices):
    float* aggU = (float*)d_ws;                          // N_U*32 f32  (1 MB)
    float* aggV = aggU + (size_t)N_U * DF;               // N_V*32 f32  (2 MB)
    float* degU = aggV + (size_t)N_V * DF;               // N_U f32
    float* degV = degU + N_U;                            // N_V f32
    _Float16* vT = (_Float16*)(degV + N_V);              // 32*N_V f16  (1 MB)
    _Float16* uT = vT + (size_t)DF * N_V;                // 32*N_U f16  (0.5 MB)
    unsigned int* bitsW = (unsigned int*)(uT + (size_t)DF * N_U); // 512*8192 u32 (16 MB)

    const size_t baseBytes =
        sizeof(float) * ((size_t)N_U * DF + (size_t)N_V * DF + N_U + N_V) +
        sizeof(_Float16) * ((size_t)DF * (N_U + N_V));
    const size_t bitsBytes = sizeof(unsigned int) * (size_t)(N_V / 32) * N_U;
    const bool useBits = ws_size >= baseBytes + bitsBytes;

    const size_t zeroBytes = sizeof(float) * ((size_t)N_U * DF + (size_t)N_V * DF + N_U + N_V);
    hipMemsetAsync(d_ws, 0, zeroBytes, stream);

    // 32*(N_V+N_U) = 786432 threads -> 3072 blocks
    convert_kernel<<<3072, 256, 0, stream>>>(u_f, v_f, vT, uT);

    // u-side: 512 row tiles (8 waves/block) x 16 K-splits; emits bitsW if room.
    agg_u_kernel<<<dim3(64, KS_A), 256, 0, stream>>>(adj, vT, aggU, degU,
                                                     useBits ? bitsW : nullptr);
    // v-side: 1024 col tiles x 8 K-splits
    if (useBits)
        agg_v_bits_kernel<<<dim3(128, KS_B), 256, 0, stream>>>(bitsW, uT, aggV, degV);
    else
        agg_v_kernel<<<dim3(128, KS_B), 256, 0, stream>>>(adj, uT, aggV, degV);

    // (8192+16384)*32 = 786432 outputs -> 3072 blocks
    finalize_kernel<<<3072, 256, 0, stream>>>(aggU, aggV, degU, degV, wU, wV, out);
}
